// BrainAgeGATv2_26929444946746
// MI455X (gfx1250) — matrix operand order repaired and checked by reading, not compiled, not run on hardware
//
#include <hip/hip_runtime.h>
#include <hip/hip_bf16.h>

// ---------------------------------------------------------------------------
// BrainAgeGATv2 on gfx1250 (MI455X): WMMA f16 GEMMs (LDS-staged A, pre-swizzled
// B fragments, fully unrolled K) + atomic edge softmax
// ---------------------------------------------------------------------------

typedef __attribute__((ext_vector_type(16))) _Float16 v16h;
typedef __attribute__((ext_vector_type(8)))  float    v8f;

union alignas(32) HFrag { v16h v; unsigned int d[8]; uint4 q[2]; _Float16 h[16]; };

#define NNODES 40000
#define NEDGES 400000
#define NB     100
#define HC     128
#define NEGS   0.2f

// ---------------- utility kernels ----------------

__global__ void k_zero_f32(float* p, int n) {
    int i = blockIdx.x * 256 + threadIdx.x;
    if (i < n) p[i] = 0.0f;
}

__global__ void k_reduce_sum(const float* __restrict__ x, int n, float* out) {
    __shared__ float sm[256];
    float v = 0.0f;
    for (int i = blockIdx.x * 256 + threadIdx.x; i < n; i += gridDim.x * 256) v += x[i];
    sm[threadIdx.x] = v;
    __syncthreads();
    for (int s = 128; s > 0; s >>= 1) {
        if (threadIdx.x < s) sm[threadIdx.x] += sm[threadIdx.x + s];
        __syncthreads();
    }
    if (threadIdx.x == 0) atomicAdd(out, sm[0]);
}

// Pack f32 weight matrix W(K,128) into f16 WMMA B-fragment order:
//   Wp[(((kt*8 + nt)*32 + lane)*16) + e] = W[(kt*32 + (lane>>4)*16 + e)*128 + nt*16 + (lane&15)]
// so each lane's 16-element fragment is 32 contiguous bytes.
__global__ void k_pack_w(const float* __restrict__ W, _Float16* __restrict__ Wp, int K) {
    int o = blockIdx.x * 256 + threadIdx.x;
    if (o >= K * 128) return;
    int e    = o & 15;
    int lane = (o >> 4) & 31;
    int tile = o >> 9;
    int nt   = tile & 7;
    int kt   = tile >> 3;
    int k    = kt * 32 + ((lane >> 4) << 4) + e;
    int col  = (nt << 4) + (lane & 15);
    Wp[o] = (_Float16)W[(size_t)k * 128 + col];
}

// ---------------- ordered-float encoding for atomic max ----------------

__device__ __forceinline__ unsigned ordenc(float f) {
    unsigned u = __float_as_uint(f);
    return (u & 0x80000000u) ? ~u : (u | 0x80000000u);
}
__device__ __forceinline__ float orddec(unsigned e) {
    return (e & 0x80000000u) ? __uint_as_float(e ^ 0x80000000u) : __uint_as_float(~e);
}

// ---------------- embedding: h0 = relu(x @ W_emb + b_emb), stored f16 ------

__global__ void k_embed(const float* __restrict__ x, const float* __restrict__ W,
                        const float* __restrict__ b, _Float16* __restrict__ h0h) {
    int i = blockIdx.x * 256 + threadIdx.x;            // N*64
    if (i >= NNODES * 64) return;
    int row = i >> 6, o = i & 63;
    const float* xr = x + row * 4;
    float v = b[o];
#pragma unroll
    for (int k = 0; k < 4; ++k) v += xr[k] * W[k * 64 + o];
    h0h[i] = (_Float16)fmaxf(v, 0.0f);
}

// ---------------- WMMA GEMM: Y(N,128) = A(N,K)f16 @ W + bias ---------------
// Block = 256 threads (8 waves); block computes 16 rows x 128 cols.
// A tile (16 x K, contiguous in global) is staged in LDS; each lane then reads
// its fragment halves as 16-byte ds loads. B comes pre-swizzled (k_pack_w) so
// each lane reads its 32-byte fragment as two b128 global loads.

template <int K>
__global__ __launch_bounds__(256) void k_wmma_gemm_bias(
    const _Float16* __restrict__ A, const _Float16* __restrict__ Wp,
    const float* __restrict__ bias, float* __restrict__ Y)
{
    __shared__ _Float16 As[16 * K];

    const int lane = threadIdx.x & 31;
    const int wave = threadIdx.x >> 5;
    const int r0   = blockIdx.x * 16;
    const int m    = lane & 15;
    const int half = lane >> 4;
    const int n    = (wave << 4) + m;

    // cooperative stage of contiguous 16*K f16 tile (8 halves / thread slot)
    {
        const uint4* src = (const uint4*)(A + (size_t)r0 * K);
        uint4* dst = (uint4*)As;
        constexpr int nvec = (16 * K) / 8;              // 16B vectors
        for (int i = threadIdx.x; i < nvec; i += 256) dst[i] = src[i];
    }
    __syncthreads();

    v8f acc = {};
#pragma unroll
    for (int kt = 0; kt < K / 32; ++kt) {
        const int k0 = kt * 32;
        HFrag a, b;
        a.q[0] = *(const uint4*)&As[m * K + k0 + half * 16];       // K = k0+half*16 .. +7
        a.q[1] = *(const uint4*)&As[m * K + k0 + half * 16 + 8];   // K = k0+half*16+8 .. +15
        const uint4* bp = (const uint4*)(Wp + (size_t)(((kt << 3) + wave) * 32 + lane) * 16);
        b.q[0] = bp[0];
        b.q[1] = bp[1];
        acc = __builtin_amdgcn_wmma_f32_16x16x32_f16(false, a.v, false, b.v,
                                                     (short)0, acc, false, false);
    }

    const float bb = bias[n];
#pragma unroll
    for (int e = 0; e < 8; ++e) {
        int row = r0 + (half << 3) + e;
        Y[(size_t)row * HC + n] = acc[e] + bb;
    }
}

// ---------------- edge kernels (2 edges per 256-thread block) --------------

__global__ __launch_bounds__(256) void k_edge_score(
    const int* __restrict__ ei, const float* __restrict__ eattr,
    const float* __restrict__ easum,
    const float* __restrict__ xl, const float* __restrict__ xr,
    const float* __restrict__ Wel, const float* __restrict__ attl,
    float* __restrict__ s, unsigned* __restrict__ smax, int E2)
{
    int t    = threadIdx.x & 127;
    int edge = blockIdx.x * 2 + (threadIdx.x >> 7);
    if (edge >= E2) return;
    int src, dst; float ea;
    if (edge < NEDGES) { src = ei[edge]; dst = ei[NEDGES + edge]; ea = eattr[edge]; }
    else               { src = dst = edge - NEDGES; ea = easum[0] * (1.0f / NEDGES); }

    float v = xl[(size_t)src * HC + t] + xr[(size_t)dst * HC + t] + ea * Wel[t];
    v = (v > 0.0f) ? v : NEGS * v;
    float p = v * attl[t];
    p += __shfl_xor(p, 1);
    p += __shfl_xor(p, 2);
    p += __shfl_xor(p, 4);
    p += __shfl_xor(p, 8);
    if ((t & 15) == 0) {
        int h = t >> 4;
        s[(size_t)edge * 8 + h] = p;
        atomicMax(&smax[(size_t)dst * 8 + h], ordenc(p));
    }
}

__global__ void k_edge_exp(const int* __restrict__ ei, float* __restrict__ s,
                           const unsigned* __restrict__ smax, float* __restrict__ den, int E2)
{
    int idx = blockIdx.x * 256 + threadIdx.x;          // E2*8
    if (idx >= E2 * 8) return;
    int edge = idx >> 3, h = idx & 7;
    int dst = (edge < NEDGES) ? ei[NEDGES + edge] : (edge - NEDGES);
    float mx = orddec(smax[(size_t)dst * 8 + h]);
    float ex = __expf(s[idx] - mx);
    s[idx] = ex;
    atomicAdd(&den[(size_t)dst * 8 + h], ex);
}

__global__ __launch_bounds__(256) void k_edge_aggregate(
    const int* __restrict__ ei, const float* __restrict__ ex,
    const float* __restrict__ den, const float* __restrict__ xl,
    float* __restrict__ out, int E2)
{
    int t    = threadIdx.x & 127;
    int edge = blockIdx.x * 2 + (threadIdx.x >> 7);
    if (edge >= E2) return;
    int src, dst;
    if (edge < NEDGES) { src = ei[edge]; dst = ei[NEDGES + edge]; }
    else               { src = dst = edge - NEDGES; }
    int h = t >> 4;
    float alpha = ex[(size_t)edge * 8 + h] / (den[(size_t)dst * 8 + h] + 1e-16f);
    atomicAdd(&out[(size_t)dst * HC + t], xl[(size_t)src * HC + t] * alpha);
}

// ---------------- batch-norm ----------------

__global__ __launch_bounds__(256) void k_bn_stats(
    const float* __restrict__ out, const float* __restrict__ bol,
    float* __restrict__ bnsum, float* __restrict__ bnsq, int rowsPerBlock)
{
    int c = threadIdx.x & 127, half = threadIdx.x >> 7;
    int r0 = blockIdx.x * rowsPerBlock;
    float bb = bol[c], s = 0.0f, s2 = 0.0f;
    for (int r = r0 + half; r < r0 + rowsPerBlock; r += 2) {
        float z = out[(size_t)r * HC + c] + bb;
        s += z; s2 += z * z;
    }
    __shared__ float sm[256], sm2[256];
    sm[threadIdx.x] = s; sm2[threadIdx.x] = s2;
    __syncthreads();
    if (half == 0) {
        atomicAdd(&bnsum[c], sm[c] + sm[c + 128]);
        atomicAdd(&bnsq[c],  sm2[c] + sm2[c + 128]);
    }
}

__global__ void k_bn_finalize(const float* bnsum, const float* bnsq,
                              float* bnmu, float* bnrs) {
    int c = threadIdx.x;
    float invN = 1.0f / NNODES;
    float mu = bnsum[c] * invN;
    float var = bnsq[c] * invN - mu * mu;
    bnmu[c] = mu;
    bnrs[c] = rsqrtf(fmaxf(var, 0.0f) + 1e-5f);
}

__global__ void k_bn_apply(const float* __restrict__ out, const float* __restrict__ bol,
                           const float* __restrict__ bnmu, const float* __restrict__ bnrs,
                           const float* __restrict__ gam, const float* __restrict__ bet,
                           float* __restrict__ h, _Float16* __restrict__ hbf, int addRes)
{
    int i = blockIdx.x * 256 + threadIdx.x;            // N*128
    if (i >= NNODES * HC) return;
    int c = i & 127;
    float z = out[i] + bol[c];
    float y = (z - bnmu[c]) * bnrs[c] * gam[c] + bet[c];
    if (addRes) y += h[i];
    y = fmaxf(y, 0.0f);
    h[i] = y;
    hbf[i] = (_Float16)y;
}

// ---------------- pooling + head ----------------

__global__ void k_pool_add(const float* __restrict__ h, const int* __restrict__ batch,
                           float* __restrict__ pooled) {
    int i = blockIdx.x * 256 + threadIdx.x;            // N*128
    if (i >= NNODES * HC) return;
    int row = i >> 7, c = i & 127;
    atomicAdd(&pooled[(size_t)batch[row] * HC + c], h[i]);
}

__global__ void k_count(const int* __restrict__ batch, float* __restrict__ cnt) {
    int i = blockIdx.x * 256 + threadIdx.x;
    if (i < NNODES) atomicAdd(&cnt[batch[i]], 1.0f);
}

__global__ __launch_bounds__(128) void k_head(
    const float* __restrict__ pooled, const float* __restrict__ cnt,
    const float* __restrict__ gfeat,
    const float* mW1, const float* mb1, const float* mW2, const float* mb2,
    const float* gW1, const float* gb1, const float* gW2, const float* gb2,
    const float* fW1, const float* fb1, const float* fW2, const float* fb2,
    const float* fW3, const float* fb3, float* __restrict__ outp)
{
    int b = blockIdx.x, t = threadIdx.x;
    __shared__ float z[160], a1[128], a2[64], m1[16], g1[16];
    const float* gf = gfeat + b * 6;
    if (t < 16) {
        float v = mb1[t];
#pragma unroll
        for (int k = 0; k < 4; ++k) v += gf[k] * mW1[k * 16 + t];
        m1[t] = fmaxf(v, 0.0f);
        float w = gb1[t];
#pragma unroll
        for (int k = 0; k < 2; ++k) w += gf[4 + k] * gW1[k * 16 + t];
        g1[t] = fmaxf(w, 0.0f);
    }
    __syncthreads();
    if (t < 16) {
        float v = mb2[t], w = gb2[t];
#pragma unroll
        for (int k = 0; k < 16; ++k) { v += m1[k] * mW2[k * 16 + t]; w += g1[k] * gW2[k * 16 + t]; }
        z[128 + t] = fmaxf(v, 0.0f);
        z[144 + t] = fmaxf(w, 0.0f);
    }
    float c = fmaxf(cnt[b], 1.0f);
    z[t] = pooled[(size_t)b * HC + t] / c;
    __syncthreads();
    float v = fb1[t];
    for (int k = 0; k < 160; ++k) v += z[k] * fW1[k * 128 + t];
    a1[t] = fmaxf(v, 0.0f);
    __syncthreads();
    if (t < 64) {
        float w = fb2[t];
        for (int k = 0; k < 128; ++k) w += a1[k] * fW2[k * 64 + t];
        a2[t] = fmaxf(w, 0.0f);
    }
    __syncthreads();
    if (t == 0) {
        float o = fb3[0];
        for (int k = 0; k < 64; ++k) o += a2[k] * fW3[k];
        outp[b] = o;
    }
}

// ---------------------------------------------------------------------------

extern "C" void kernel_launch(void* const* d_in, const int* in_sizes, int n_in,
                              void* d_out, int out_size, void* d_ws, size_t ws_size,
                              hipStream_t stream) {
    const float* x        = (const float*)d_in[0];
    const int*   ei       = (const int*)d_in[1];
    const float* eattr    = (const float*)d_in[2];
    const int*   batch    = (const int*)d_in[3];
    const float* gfeat    = (const float*)d_in[4];
    const float* W_emb    = (const float*)d_in[5];
    const float* b_emb    = (const float*)d_in[6];
    const float* Wl1      = (const float*)d_in[7];
    const float* Wr1      = (const float*)d_in[8];
    const float* Wl234    = (const float*)d_in[9];
    const float* Wr234    = (const float*)d_in[10];
    const float* bl       = (const float*)d_in[11];
    const float* br       = (const float*)d_in[12];
    const float* We       = (const float*)d_in[13];
    const float* att      = (const float*)d_in[14];
    const float* bo       = (const float*)d_in[15];
    const float* gamma    = (const float*)d_in[16];
    const float* beta     = (const float*)d_in[17];
    const float* mW1      = (const float*)d_in[18];
    const float* mb1      = (const float*)d_in[19];
    const float* mW2      = (const float*)d_in[20];
    const float* mb2      = (const float*)d_in[21];
    const float* gW1      = (const float*)d_in[22];
    const float* gb1      = (const float*)d_in[23];
    const float* gW2      = (const float*)d_in[24];
    const float* gb2      = (const float*)d_in[25];
    const float* fW1      = (const float*)d_in[26];
    const float* fb1      = (const float*)d_in[27];
    const float* fW2      = (const float*)d_in[28];
    const float* fb2      = (const float*)d_in[29];
    const float* fW3      = (const float*)d_in[30];
    const float* fb3      = (const float*)d_in[31];
    float* outp = (float*)d_out;

    const int E2 = NEDGES + NNODES;

    // --- workspace carve ---
    size_t cur = 0;
    auto take = [&](size_t bytes) -> char* {
        size_t o = (cur + 255) & ~(size_t)255;
        cur = o + bytes;
        return (char*)d_ws + o;
    };
    float*     h      = (float*)take((size_t)NNODES * HC * 4);
    float*     xl     = (float*)take((size_t)NNODES * HC * 4);
    float*     xr     = (float*)take((size_t)NNODES * HC * 4);
    float*     out_ac = (float*)take((size_t)NNODES * HC * 4);
    float*     s      = (float*)take((size_t)E2 * 8 * 4);
    unsigned*  smax   = (unsigned*)take((size_t)NNODES * 8 * 4);
    float*     den    = (float*)take((size_t)NNODES * 8 * 4);
    _Float16*  hbf    = (_Float16*)take((size_t)NNODES * HC * 2);
    _Float16*  h0h    = (_Float16*)take((size_t)NNODES * 64 * 2);
    _Float16*  wl1h   = (_Float16*)take((size_t)64 * 128 * 2);
    _Float16*  wr1h   = (_Float16*)take((size_t)64 * 128 * 2);
    _Float16*  wlh    = (_Float16*)take((size_t)3 * 128 * 128 * 2);
    _Float16*  wrh    = (_Float16*)take((size_t)3 * 128 * 128 * 2);
    float*     bnsum  = (float*)take(128 * 4);
    float*     bnsq   = (float*)take(128 * 4);
    float*     bnmu   = (float*)take(128 * 4);
    float*     bnrs   = (float*)take(128 * 4);
    float*     pooled = (float*)take((size_t)NB * HC * 4);
    float*     cnt    = (float*)take(NB * 4);
    float*     easum  = (float*)take(4);
    (void)ws_size; (void)n_in; (void)in_sizes; (void)out_size;

    auto zero = [&](float* p, int n) {
        k_zero_f32<<<(n + 255) / 256, 256, 0, stream>>>(p, n);
    };

    // --- one-time-per-call setup ---
    zero(easum, 1);
    zero(pooled, NB * HC);
    zero(cnt, NB);
    k_reduce_sum<<<256, 256, 0, stream>>>(eattr, NEDGES, easum);
    k_pack_w<<<(64 * 128 + 255) / 256, 256, 0, stream>>>(Wl1, wl1h, 64);
    k_pack_w<<<(64 * 128 + 255) / 256, 256, 0, stream>>>(Wr1, wr1h, 64);
    for (int i = 0; i < 3; ++i) {
        k_pack_w<<<(128 * 128 + 255) / 256, 256, 0, stream>>>(
            Wl234 + (size_t)i * 128 * 128, wlh + (size_t)i * 128 * 128, 128);
        k_pack_w<<<(128 * 128 + 255) / 256, 256, 0, stream>>>(
            Wr234 + (size_t)i * 128 * 128, wrh + (size_t)i * 128 * 128, 128);
    }
    k_embed<<<(NNODES * 64 + 255) / 256, 256, 0, stream>>>(x, W_emb, b_emb, h0h);
    k_count<<<(NNODES + 255) / 256, 256, 0, stream>>>(batch, cnt);

    // --- 4 GAT layers ---
    for (int li = 0; li < 4; ++li) {
        const _Float16* A  = (li == 0) ? h0h : hbf;
        const _Float16* WL = (li == 0) ? wl1h : (wlh + (size_t)(li - 1) * 128 * 128);
        const _Float16* WR = (li == 0) ? wr1h : (wrh + (size_t)(li - 1) * 128 * 128);

        if (li == 0) {
            k_wmma_gemm_bias<64><<<NNODES / 16, 256, 0, stream>>>(A, WL, bl + li * HC, xl);
            k_wmma_gemm_bias<64><<<NNODES / 16, 256, 0, stream>>>(A, WR, br + li * HC, xr);
        } else {
            k_wmma_gemm_bias<128><<<NNODES / 16, 256, 0, stream>>>(A, WL, bl + li * HC, xl);
            k_wmma_gemm_bias<128><<<NNODES / 16, 256, 0, stream>>>(A, WR, br + li * HC, xr);
        }

        zero((float*)smax, NNODES * 8);
        zero(den, NNODES * 8);
        zero(out_ac, NNODES * HC);
        zero(bnsum, 128);
        zero(bnsq, 128);

        k_edge_score<<<E2 / 2, 256, 0, stream>>>(ei, eattr, easum, xl, xr,
                                                 We + li * HC, att + li * HC, s, smax, E2);
        k_edge_exp<<<(E2 * 8 + 255) / 256, 256, 0, stream>>>(ei, s, smax, den, E2);
        k_edge_aggregate<<<E2 / 2, 256, 0, stream>>>(ei, s, den, xl, out_ac, E2);

        k_bn_stats<<<NNODES / 80, 256, 0, stream>>>(out_ac, bo + li * HC, bnsum, bnsq, 80);
        k_bn_finalize<<<1, 128, 0, stream>>>(bnsum, bnsq, bnmu, bnrs);
        k_bn_apply<<<(NNODES * HC + 255) / 256, 256, 0, stream>>>(
            out_ac, bo + li * HC, bnmu, bnrs, gamma + li * HC, beta + li * HC,
            h, hbf, (li == 0) ? 0 : 1);
    }

    // --- pooling + head ---
    k_pool_add<<<(NNODES * HC + 255) / 256, 256, 0, stream>>>(h, batch, pooled);
    k_head<<<NB, 128, 0, stream>>>(pooled, cnt, gfeat,
                                   mW1, mb1, mW2, mb2, gW1, gb1, gW2, gb2,
                                   fW1, fb1, fW2, fb2, fW3, fb3, outp);
}
